// IBIGraphClassifier_40278203302581
// MI455X (gfx1250) — compile-verified, matrix operand order checked by reading, not run on hardware
//
#include <hip/hip_runtime.h>
#include <hip/hip_bf16.h>
#include <math.h>

#define B_  64
#define N_  512
#define F_  64
#define D_  256
#define NHEAD_ 4
#define DH_ 64
#define TOK (B_*N_)
#define NEGV (-1e9f)

typedef __attribute__((ext_vector_type(16))) __bf16          v16bf;
typedef __attribute__((ext_vector_type(8)))  unsigned        v8u;
typedef __attribute__((ext_vector_type(8)))  float           v8f;

__device__ __forceinline__ unsigned short f2bf(float f){
  unsigned int u = __builtin_bit_cast(unsigned int, f);
  u += 0x7FFFu + ((u >> 16) & 1u);              // round-to-nearest-even
  return (unsigned short)(u >> 16);
}
// pack two f32 -> bf16x2 in one VALU op when the HW conversion exists
__device__ __forceinline__ unsigned pkbf(float a, float b){
#if __has_builtin(__builtin_amdgcn_cvt_pk_bf16_f32)
  auto p = __builtin_amdgcn_cvt_pk_bf16_f32(a, b);
  return __builtin_bit_cast(unsigned, p);
#else
  return (unsigned)f2bf(a) | ((unsigned)f2bf(b) << 16);
#endif
}

// WMMA 16x16x32 bf16 lane mapping (ISA 7.12.2):
// lane L: sel = L&15, kh = L>>4; element i=2e+t <-> k = kh*8 + (e>=4?16:0) + (e&3)*2 + t
// Inverse (for swizzled LDS staging): lane = sel + 16*((k>>3)&1), slot i = (k>>4)*8 + (k&7)
__device__ __forceinline__ v16bf frag_f32(const float* p0, int ss, int sk){
  int lane = threadIdx.x & 31;
  const float* p = p0 + (lane & 15) * ss + (lane >> 4) * 8 * sk;
  v8u r;
#pragma unroll
  for (int e = 0; e < 8; ++e){
    int kb = ((e >= 4) ? 16 : 0) + (e & 3) * 2;
    r[e] = pkbf(p[kb * sk], p[(kb + 1) * sk]);
  }
  return __builtin_bit_cast(v16bf, r);
}
__device__ __forceinline__ v8f wmma_bf16(v16bf a, v16bf b, v8f c){
  return __builtin_amdgcn_wmma_f32_16x16x32_bf16(false, a, false, b, (short)0, c, false, false);
}
__device__ __forceinline__ float actf(float x, int act){
  if (act == 1) return 0.5f * x * (1.0f + erff(x * 0.70710678118f)); // exact gelu
  if (act == 2) return tanhf(x);
  if (act == 3) return 1.0f / (1.0f + __expf(-x));
  return x;
}

// ---------------- generic (optionally batched) GEMM ----------------
// out = actf(A@op(W) + bias) + addend.  A:[M,K](lda)
// transW==0: W is [K,N] with row stride ldw.  transW==1: W is [N,K] with row
// stride ldw (i.e. B = W^T) -- used for X X^T similarity.
// Block tile 128x64, 8 waves, each wave 32x32 (2x2 wmma accumulators).
// Register-double-buffered staging: next chunk's global loads issue under the
// current chunk's WMMAs. LDS is laid out in fragment order so a fragment load
// is one contiguous 32B LDS read per lane.
__global__ __launch_bounds__(256) void gemm_k(const float* __restrict__ A,
    const float* __restrict__ W, const float* __restrict__ bias,
    const float* __restrict__ addend, float* __restrict__ out,
    int M, int K, int N, int lda, int ldw, int ldc, int act, int transW,
    long bsA, long bsW, long bsO){
  __shared__ __align__(32) unsigned short sA[8*512];   // 8 subtiles of 16x32
  __shared__ __align__(32) unsigned short sW[4*512];   // 4 subtiles of 32x16
  int bz = blockIdx.z;
  A += (size_t)bz * bsA;  W += (size_t)bz * bsW;
  float* outp = out + (size_t)bz * bsO;
  const float* addp = addend ? addend + (size_t)bz * bsO : nullptr;
  int m0 = blockIdx.y * 128, n0 = blockIdx.x * 64;
  int t = threadIdx.x, w = t >> 5, lane = t & 31;
  int wy = w >> 1, wx = w & 1;                          // 4x2 wave grid
  v8f acc[2][2] = {};
  const float4 z4 = make_float4(0.f, 0.f, 0.f, 0.f);
  float4 ra[4], rw[2];

  auto loadA = [&](int k0){
#pragma unroll
    for (int it = 0; it < 4; ++it){
      int idx = t + it*256;                 // 1024 groups of 4 consecutive k
      int r = idx >> 3, c4 = (idx & 7) * 4;
      int m = m0 + r;
      ra[it] = (m < M) ? *(const float4*)(A + (size_t)m*lda + k0 + c4) : z4;
    }
  };
  auto loadW = [&](int k0){
#pragma unroll
    for (int it = 0; it < 2; ++it){
      int idx = t + it*256;                 // 512 groups: (4 consecutive k) x n
      int n = idx & 63, k = (idx >> 6) * 4;
      int nn = n0 + n;
      if (nn < N){
        if (transW){
          rw[it] = *(const float4*)(W + (size_t)nn*ldw + k0 + k);
        } else {
          const float* wp = W + (size_t)(k0 + k)*ldw + nn;
          rw[it] = make_float4(wp[0], wp[(size_t)ldw],
                               wp[2*(size_t)ldw], wp[3*(size_t)ldw]);
        }
      } else rw[it] = z4;
    }
  };
  auto stage = [&](){
#pragma unroll
    for (int it = 0; it < 4; ++it){
      int idx = t + it*256;
      int r = idx >> 3, c4 = (idx & 7) * 4;
      unsigned lds = (unsigned)((r >> 4)*512
                   + ((r & 15) + ((c4 >> 3) & 1)*16)*16
                   + ((c4 >> 4)*8 + (c4 & 7)));
      *(uint2*)(sA + lds) = make_uint2(pkbf(ra[it].x, ra[it].y),
                                       pkbf(ra[it].z, ra[it].w));
    }
#pragma unroll
    for (int it = 0; it < 2; ++it){
      int idx = t + it*256;
      int n = idx & 63, k = (idx >> 6) * 4;
      unsigned lds = (unsigned)((n >> 4)*512
                   + ((n & 15) + ((k >> 3) & 1)*16)*16
                   + ((k >> 4)*8 + (k & 7)));
      *(uint2*)(sW + lds) = make_uint2(pkbf(rw[it].x, rw[it].y),
                                       pkbf(rw[it].z, rw[it].w));
    }
  };

  loadA(0); loadW(0);
  for (int k0 = 0; k0 < K; k0 += 32){
    stage();
    __syncthreads();
    v16bf a0 = *(const v16bf*)(sA + (wy*2+0)*512 + lane*16);
    v16bf a1 = *(const v16bf*)(sA + (wy*2+1)*512 + lane*16);
    v16bf b0 = *(const v16bf*)(sW + (wx*2+0)*512 + lane*16);
    v16bf b1 = *(const v16bf*)(sW + (wx*2+1)*512 + lane*16);
    if (k0 + 32 < K){
      if (k0 + 64 < K){                     // prefetch 2 chunks ahead
        int r = t >> 3, c4 = (t & 7) * 4;
        int m = m0 + r;
        if (m < M) __builtin_prefetch(A + (size_t)m*lda + k0 + 64 + c4, 0, 0);
      }
      loadA(k0 + 32); loadW(k0 + 32);       // overlap with WMMAs below
    }
    acc[0][0] = wmma_bf16(a0, b0, acc[0][0]);
    acc[0][1] = wmma_bf16(a0, b1, acc[0][1]);
    acc[1][0] = wmma_bf16(a1, b0, acc[1][0]);
    acc[1][1] = wmma_bf16(a1, b1, acc[1][1]);
    __syncthreads();
  }
  int nl = lane & 15, mh = (lane >> 4) * 8;
#pragma unroll
  for (int tj = 0; tj < 2; ++tj){
    int n = n0 + wx*32 + tj*16 + nl;
    if (n >= N) continue;
    float bv = bias ? bias[n] : 0.0f;
#pragma unroll
    for (int ti = 0; ti < 2; ++ti){
#pragma unroll
      for (int r = 0; r < 8; ++r){
        int m = m0 + wy*32 + ti*16 + mh + r;
        if (m < M){
          float vv = actf(acc[ti][tj][r] + bv, act);
          if (addp) vv += addp[(size_t)m*ldc + n];
          outp[(size_t)m*ldc + n] = vv;
        }
      }
    }
  }
}

// ---------------- LayerNorm (wave per row), optional fused GELU ----------------
__global__ __launch_bounds__(256) void ln_k(const float* __restrict__ in,
    const float* __restrict__ g, const float* __restrict__ b,
    float* __restrict__ out, int R, int D, int gelu){
  int w = threadIdx.x >> 5, lane = threadIdx.x & 31;
  int row = blockIdx.x * 8 + w;
  if (row >= R) return;
  const float* x = in + (size_t)row * D;
  float s = 0.f;
  for (int j = lane; j < D; j += 32) s += x[j];
  for (int off = 16; off; off >>= 1) s += __shfl_xor(s, off);
  float mu = s / D;
  float v = 0.f;
  for (int j = lane; j < D; j += 32){ float d = x[j]-mu; v += d*d; }
  for (int off = 16; off; off >>= 1) v += __shfl_xor(v, off);
  float inv = rsqrtf(v / D + 1e-5f);
  float* o = out + (size_t)row * D;
  for (int j = lane; j < D; j += 32){
    float y = (x[j]-mu) * inv * g[j] + b[j];
    if (gelu) y = 0.5f*y*(1.0f+erff(y*0.70710678118f));
    o[j] = y;
  }
}

// ---------------- fused MHA: per (q-tile16, head, batch) ----------------
#define SLD 520
__global__ __launch_bounds__(256) void attn_k(const float* __restrict__ Q,
    const float* __restrict__ Km, const float* __restrict__ Vm,
    const unsigned char* __restrict__ vm, float* __restrict__ O){
  __shared__ float S[16*SLD];
  int q0 = blockIdx.x * 16, h = blockIdx.y, b = blockIdx.z;
  int t = threadIdx.x, w = t >> 5, lane = t & 31;
  const float* qb = Q  + (size_t)(b*N_)*D_ + h*DH_;
  const float* kb = Km + (size_t)(b*N_)*D_ + h*DH_;
  const float* vb = Vm + (size_t)(b*N_)*D_ + h*DH_;
  // scores = Q K^T / sqrt(dh), key-masked ; Q fragments are kt-invariant: hoist
  v16bf fq0 = frag_f32(qb + (size_t)q0*D_ +  0, D_, 1);
  v16bf fq1 = frag_f32(qb + (size_t)q0*D_ + 32, D_, 1);
  for (int kt = w; kt < N_/16; kt += 8){
    v8f acc = {};
    acc = wmma_bf16(fq0, frag_f32(kb + (size_t)(kt*16)*D_ +  0, D_, 1), acc);
    acc = wmma_bf16(fq1, frag_f32(kb + (size_t)(kt*16)*D_ + 32, D_, 1), acc);
    int j = kt*16 + (lane & 15);
    bool ok = vm[b*N_ + j] != 0;
    int mh = (lane >> 4) * 8;
#pragma unroll
    for (int r = 0; r < 8; ++r)
      S[(mh + r)*SLD + j] = ok ? acc[r] * 0.125f : NEGV;
  }
  __syncthreads();
  // row softmax (wave handles 2 rows)
  for (int rr = 0; rr < 2; ++rr){
    float* sr = S + (w*2 + rr)*SLD;
    float mx = NEGV;
    for (int j = lane; j < N_; j += 32) mx = fmaxf(mx, sr[j]);
    for (int off = 16; off; off >>= 1) mx = fmaxf(mx, __shfl_xor(mx, off));
    float sm = 0.f;
    for (int j = lane; j < N_; j += 32){ float e = __expf(sr[j]-mx); sr[j] = e; sm += e; }
    for (int off = 16; off; off >>= 1) sm += __shfl_xor(sm, off);
    float inv = 1.0f / sm;
    for (int j = lane; j < N_; j += 32) sr[j] *= inv;
  }
  __syncthreads();
  // O = P @ V  (waves 0..3 cover dh=64)
  if (w < 4){
    v8f acc = {};
    for (int kt = 0; kt < N_/32; ++kt){
      v16bf fp = frag_f32(S + kt*32, SLD, 1);
      v16bf fv = frag_f32(vb + (size_t)(kt*32)*D_ + w*16, 1, D_);
      acc = wmma_bf16(fp, fv, acc);
    }
    int n = lane & 15, mh = (lane >> 4) * 8;
#pragma unroll
    for (int r = 0; r < 8; ++r)
      O[(size_t)(b*N_ + q0 + mh + r)*D_ + h*DH_ + w*16 + n] = acc[r];
  }
}

// ---------------- mask raw similarity: invalid / diagonal -> NEG ----------------
__global__ void simmask_k(float* __restrict__ Sm, const unsigned char* __restrict__ vm){
  size_t i = (size_t)blockIdx.x*256 + threadIdx.x;
  if (i >= (size_t)B_*N_*N_) return;
  int jj = (int)(i % N_);
  int ii = (int)((i / N_) % N_);
  int b  = (int)(i / ((size_t)N_*N_));
  bool ok = (vm[b*N_+ii] != 0) && (vm[b*N_+jj] != 0) && (ii != jj);
  if (!ok) Sm[i] = NEGV;
}

// ---------------- per-row top-8 threshold -> 0/1 adjacency ----------------
__global__ __launch_bounds__(256) void topk_k(const float* __restrict__ Sm,
                                              float* __restrict__ A){
  int w = threadIdx.x >> 5, lane = threadIdx.x & 31;
  int row = blockIdx.x * 8 + w;
  const float* s = Sm + (size_t)row * N_;
  float vals[16];
#pragma unroll
  for (int q = 0; q < 16; ++q) vals[q] = s[lane + 32*q];
  unsigned rem = 0;
  float thr = NEGV;
  for (int it = 0; it < 8; ++it){
    float lm = -3.0e38f; int ls = -1;
#pragma unroll
    for (int q = 0; q < 16; ++q)
      if (!((rem >> q) & 1u) && vals[q] > lm){ lm = vals[q]; ls = q; }
    float bm = lm; int bl = lane;
    for (int off = 16; off; off >>= 1){
      float ov = __shfl_xor(bm, off);
      int  ol = __shfl_xor(bl, off);
      if (ov > bm || (ov == bm && ol < bl)){ bm = ov; bl = ol; }
    }
    if (bl == lane && ls >= 0) rem |= 1u << ls;
    thr = bm;
  }
  float* a = A + (size_t)row * N_;
#pragma unroll
  for (int q = 0; q < 16; ++q) a[lane + 32*q] = (vals[q] >= thr) ? 1.0f : 0.0f;
}

// ---------------- A2 = max(A, A^T) + hop(1,2) edges + diagonal ----------------
__global__ void symhop_k(const float* __restrict__ A,
                         const unsigned char* __restrict__ vm,
                         float* __restrict__ A2){
  size_t i = (size_t)blockIdx.x*256 + threadIdx.x;
  if (i >= (size_t)B_*N_*N_) return;
  int jj = (int)(i % N_);
  int ii = (int)((i / N_) % N_);
  int b  = (int)(i / ((size_t)N_*N_));
  float v = fmaxf(A[i], A[((size_t)b*N_ + jj)*N_ + ii]);
  bool vi = vm[b*N_+ii] != 0, vj = vm[b*N_+jj] != 0;
  int d = ii - jj; if (d < 0) d = -d;
  if (vi && vj && d >= 1 && d <= 2) v = fmaxf(v, 1.0f);
  if (ii == jj && vi)               v = fmaxf(v, 1.0f);
  A2[i] = v;
}

// ---------------- small elementwise / reduction kernels ----------------
__global__ void addpos_k(float* x, const float* pos){
  size_t i = (size_t)blockIdx.x*256 + threadIdx.x;
  if (i >= (size_t)TOK*D_) return;
  x[i] += pos[i % ((size_t)N_*D_)];
}
__global__ void gatescale_k(float* A2, const float* gate){
  size_t i = (size_t)blockIdx.x*256 + threadIdx.x;
  if (i >= (size_t)B_*N_*N_) return;
  int jj = (int)(i % N_);
  int b  = (int)(i / ((size_t)N_*N_));
  A2[i] *= gate[b*N_ + jj];
}
__global__ __launch_bounds__(256) void deg_k(const float* A2, float* deg){
  int w = threadIdx.x >> 5, lane = threadIdx.x & 31;
  int row = blockIdx.x * 8 + w;
  const float* a = A2 + (size_t)row * N_;
  float s = 0.f;
  for (int j = lane; j < N_; j += 32) s += a[j];
  for (int off = 16; off; off >>= 1) s += __shfl_xor(s, off);
  if (lane == 0) deg[row] = fmaxf(s, 1.0f);
}
__global__ void rowdiv_k(float* x, const float* deg){
  size_t i = (size_t)blockIdx.x*256 + threadIdx.x;
  if (i >= (size_t)TOK*D_) return;
  x[i] /= deg[i / D_];
}
__global__ void catcopy_k(const float* src, float* dst){
  size_t i = (size_t)blockIdx.x*256 + threadIdx.x;
  if (i >= (size_t)TOK*D_) return;
  size_t row = i / D_, d = i % D_;
  dst[row*512 + 256 + d] = src[i];
}
__global__ void maskrows_k(float* x, const unsigned char* vm){
  size_t i = (size_t)blockIdx.x*256 + threadIdx.x;
  if (i >= (size_t)TOK*D_) return;
  if (!vm[i / D_]) x[i] = 0.0f;
}
__global__ __launch_bounds__(256) void rownorm_k(const float* x, float* xn){
  int w = threadIdx.x >> 5, lane = threadIdx.x & 31;
  int row = blockIdx.x * 8 + w;
  const float* xr = x + (size_t)row * D_;
  float s = 0.f;
  for (int j = lane; j < D_; j += 32){ float v = xr[j]; s += v*v; }
  for (int off = 16; off; off >>= 1) s += __shfl_xor(s, off);
  float inv = 1.0f / fmaxf(sqrtf(s), 1e-12f);
  float* o = xn + (size_t)row * D_;
  for (int j = lane; j < D_; j += 32) o[j] = xr[j] * inv;
}
__global__ __launch_bounds__(256) void meanmax_k(const float* x,
    const unsigned char* vm, float* g){
  int b = blockIdx.x, d = threadIdx.x;     // 256 threads == D_
  float s = 0.f, mx = NEGV, cnt = 0.f;
  for (int n = 0; n < N_; ++n){
    if (vm[b*N_ + n]){
      float v = x[(size_t)(b*N_ + n)*D_ + d];
      s += v; mx = fmaxf(mx, v); cnt += 1.0f;
    }
  }
  g[(size_t)b*768 + d]       = s / fmaxf(cnt, 1.0f);
  g[(size_t)b*768 + 256 + d] = mx;
}
__global__ __launch_bounds__(256) void poolattn_k(const float* x, const float* sc,
    const unsigned char* vm, float* out){   // out base = P4 + h*256, row stride 1024
  __shared__ float red[2];
  int b = blockIdx.x, t = threadIdx.x;
  if (t < 32){
    float mx = NEGV;
    for (int n = t; n < N_; n += 32)
      mx = fmaxf(mx, vm[b*N_+n] ? sc[b*N_+n] : NEGV);
    for (int off = 16; off; off >>= 1) mx = fmaxf(mx, __shfl_xor(mx, off));
    float sm = 0.f;
    for (int n = t; n < N_; n += 32)
      sm += __expf((vm[b*N_+n] ? sc[b*N_+n] : NEGV) - mx);
    for (int off = 16; off; off >>= 1) sm += __shfl_xor(sm, off);
    if (t == 0){ red[0] = mx; red[1] = sm; }
  }
  __syncthreads();
  float mx = red[0], inv = 1.0f / red[1];
  float acc = 0.f;
  for (int n = 0; n < N_; ++n){
    float wgt = __expf((vm[b*N_+n] ? sc[b*N_+n] : NEGV) - mx) * inv;
    acc += wgt * x[(size_t)(b*N_ + n)*D_ + t];
  }
  out[(size_t)b*1024 + t] = acc;
}

// =======================================================================
extern "C" void kernel_launch(void* const* d_in, const int* in_sizes, int n_in,
                              void* d_out, int out_size, void* d_ws, size_t ws_size,
                              hipStream_t stream){
  const float* beats = (const float*)d_in[0];
  const unsigned char* vm = (const unsigned char*)d_in[2]; // bool mask [B,N]
  auto P = [&](int i){ return (const float*)d_in[i]; };

  // ---- workspace layout (floats) ----
  float* ws = (float*)d_ws;
  size_t o = 0;
  float* X   = ws + o; o += (size_t)TOK*D_;
  float* T0  = ws + o; o += (size_t)TOK*D_;
  float* Qb  = ws + o; o += (size_t)TOK*D_;
  float* Kb  = ws + o; o += (size_t)TOK*D_;
  float* Vb  = ws + o; o += (size_t)TOK*D_;
  float* Ob  = ws + o; o += (size_t)TOK*D_;
  float* FF  = ws + o; o += (size_t)TOK*1024;   // FFN; GNN: SIM | Adj
  float* CAT = ws + o; o += (size_t)TOK*512;    // enc h1; GNN: A2/Aeff then concat
  float* G1  = ws + o; o += (size_t)TOK*512;    // enc h2; u1 out; pool h1 out
  float* SC  = ws + o; o += TOK;                // gate / pooling scores
  float* DEG = ws + o; o += TOK;
  float* P4  = ws + o; o += (size_t)B_*1024;
  float* Gg  = ws + o; o += (size_t)B_*768;
  float* Z1  = ws + o; o += (size_t)B_*512;
  float* Z2  = ws + o; o += (size_t)B_*256;
  float* SIM = FF;
  float* Adj = FF + (size_t)TOK*512;
  float* A2  = CAT;

  auto GEMM = [&](const float* A, const float* W, const float* bias,
                  const float* addend, float* out, int M, int K, int N,
                  int lda, int ldc, int act){
    dim3 g((N + 63)/64, (M + 127)/128, 1);
    gemm_k<<<g, 256, 0, stream>>>(A, W, bias, addend, out, M, K, N,
                                  lda, N, ldc, act, 0, 0, 0, 0);
  };
  // batched: out[b] = Ab[b] (NxN) @ Xin[b] (NxD)
  auto BGEMM = [&](const float* Ab, const float* Xin, float* out){
    dim3 g((D_ + 63)/64, (N_ + 127)/128, B_);
    gemm_k<<<g, 256, 0, stream>>>(Ab, Xin, nullptr, nullptr, out,
                                  N_, N_, D_, N_, D_, D_, 0, 0,
                                  (long)N_*N_, (long)N_*D_, (long)N_*D_);
  };
  // batched similarity: Sm[b] = Xn[b] @ Xn[b]^T  (transposed-B path)
  auto SIMGEMM = [&](const float* Xn, float* Sm){
    dim3 g((N_ + 63)/64, (N_ + 127)/128, B_);
    gemm_k<<<g, 256, 0, stream>>>(Xn, Xn, nullptr, nullptr, Sm,
                                  N_, D_, N_, D_, D_, N_, 0, 1,
                                  (long)N_*D_, (long)N_*D_, (long)N_*N_);
  };
  auto LN = [&](const float* in, int gi, float* out, int R, int D, int gelu){
    ln_k<<<(R + 7)/8, 256, 0, stream>>>(in, P(gi), P(gi+1), out, R, D, gelu);
  };
  const int EW = (TOK*D_)/256;       // elementwise grid over [TOK,256]
  const int EA = (B_*N_*N_)/256;     // elementwise grid over [B,N,N]

  // ---------------- encoder ----------------
  GEMM(beats, P(3),  P(4),  nullptr, CAT, TOK, F_,  512, F_,  512, 0);  // l1
  LN(CAT, 5, CAT, TOK, 512, 1);                                         // n1+gelu
  GEMM(CAT,   P(7),  P(8),  nullptr, G1,  TOK, 512, 512, 512, 512, 0);  // l2
  LN(G1, 9, G1, TOK, 512, 1);                                           // n2+gelu
  GEMM(beats, P(13), P(14), nullptr, X,   TOK, F_,  D_,  F_,  D_,  0);  // res
  GEMM(G1,    P(11), P(12), X,       T0,  TOK, 512, D_,  512, D_,  0);  // l3 + res
  LN(T0, 15, X, TOK, D_, 0);                                            // norm
  addpos_k<<<EW, 256, 0, stream>>>(X, P(17));                           // + pos

  // ---------------- transformer layers ----------------
  for (int l = 0; l < 2; ++l){
    int p = 18 + 16*l;
    LN(X, p+0, T0, TOK, D_, 0);                                         // ln1
    GEMM(T0, P(p+2), P(p+3), nullptr, Qb, TOK, D_, D_, D_, D_, 0);      // wq
    GEMM(T0, P(p+4), P(p+5), nullptr, Kb, TOK, D_, D_, D_, D_, 0);      // wk
    GEMM(T0, P(p+6), P(p+7), nullptr, Vb, TOK, D_, D_, D_, D_, 0);      // wv
    attn_k<<<dim3(N_/16, NHEAD_, B_), 256, 0, stream>>>(Qb, Kb, Vb, vm, Ob);
    GEMM(Ob, P(p+8), P(p+9), X, X, TOK, D_, D_, D_, D_, 0);             // wo + res
    LN(X, p+10, T0, TOK, D_, 0);                                        // ln2
    GEMM(T0, P(p+12), P(p+13), nullptr, FF, TOK, D_, 1024, D_, 1024, 1);// f1+gelu
    GEMM(FF, P(p+14), P(p+15), X, X, TOK, 1024, D_, 1024, D_, 0);       // f2 + res
  }

  // ---------------- GNN layers ----------------
  for (int l = 0; l < 3; ++l){
    int p = 50 + 16*l;
    rownorm_k<<<TOK/8, 256, 0, stream>>>(X, T0);                        // xn
    SIMGEMM(T0, SIM);                                                   // Xn Xn^T
    simmask_k<<<EA, 256, 0, stream>>>(SIM, vm);                         // mask/eye
    topk_k<<<TOK/8, 256, 0, stream>>>(SIM, Adj);                        // top-8 thr
    symhop_k<<<EA, 256, 0, stream>>>(Adj, vm, A2);                      // sym + hops
    GEMM(X, P(p+6), P(p+7), nullptr, Qb, TOK, D_, D_, D_, D_, 0);       // xg = x@gd
    BGEMM(A2, Qb, Vb);                                                  // A@xg
    GEMM(X, P(p+4), P(p+5), Vb, Kb, TOK, D_, D_, D_, D_, 0);            // + x@gs
    GEMM(Kb, P(p+8), P(p+9), nullptr, SC, TOK, D_, 1, D_, 1, 3);        // gate
    gatescale_k<<<EA, 256, 0, stream>>>(A2, SC);                        // Aeff
    deg_k<<<TOK/8, 256, 0, stream>>>(A2, DEG);                          // deg
    GEMM(X, P(p+0), P(p+1), nullptr, Qb, TOK, D_, D_, D_, D_, 0);       // msg
    BGEMM(A2, Qb, Ob);                                                  // Aeff@msg
    rowdiv_k<<<EW, 256, 0, stream>>>(Ob, DEG);                          // /deg
    GEMM(X, P(p+2), P(p+3), nullptr, CAT, TOK, D_, D_, D_, 512, 0);     // slf -> [:,0:256]
    catcopy_k<<<EW, 256, 0, stream>>>(Ob, CAT);                         // agg -> [:,256:512]
    GEMM(CAT, P(p+10), P(p+11), nullptr, G1, TOK, 512, 512, 512, 512, 1);// u1+gelu
    GEMM(G1,  P(p+12), P(p+13), X,       T0, TOK, 512, D_,  512, D_,  0);// u2 + res
    LN(T0, p+14, X, TOK, D_, 0);                                        // norm
    maskrows_k<<<EW, 256, 0, stream>>>(X, vm);                          // * mask
  }

  // ---------------- pooling + head ----------------
  meanmax_k<<<B_, 256, 0, stream>>>(X, vm, Gg);                         // mean|max
  for (int h = 0; h < 4; ++h){
    int p = 98 + 4*h;
    GEMM(X,  P(p+0), P(p+1), nullptr, G1, TOK, D_, 128, D_, 128, 2);    // tanh(x@h1)
    GEMM(G1, P(p+2), P(p+3), nullptr, SC, TOK, 128, 1, 128, 1, 0);      // scores
    poolattn_k<<<B_, 256, 0, stream>>>(X, SC, vm, P4 + h*256);          // weighted sum
  }
  GEMM(P4, P(114), P(115), nullptr, Gg + 512, B_, 1024, D_, 1024, 768, 0); // pool out
  GEMM(Gg, P(116), P(117), nullptr, Z1, B_, 768, 512, 768, 512, 0);     // c1
  LN(Z1, 118, Z1, B_, 512, 1);                                          // n1+gelu
  GEMM(Z1, P(120), P(121), nullptr, Z2, B_, 512, D_, 512, D_, 1);       // c2+gelu
  GEMM(Z2, P(122), P(123), nullptr, (float*)d_out, B_, D_, 5, D_, 5, 0);// c3 -> out
  (void)in_sizes; (void)n_in; (void)out_size; (void)ws_size;
}
